// FocalUFormerBlock_69818988363883
// MI455X (gfx1250) — matrix operand order repaired and checked by reading, not compiled, not run on hardware
//
#include <hip/hip_runtime.h>
#include <math.h>

typedef __bf16 bf16;
typedef __attribute__((ext_vector_type(16))) __bf16 v16bf;
typedef __attribute__((ext_vector_type(8)))  float   v8f;
typedef unsigned int u32x4 __attribute__((ext_vector_type(4)));
typedef int          i32x8 __attribute__((ext_vector_type(8)));
typedef int          i32x4 __attribute__((ext_vector_type(4)));

#define NT   36864   // tokens (192*192)
#define IMG  192
#define CCH  128
#define LTOT 436     // 256 + 144 + 36
#define LPAD 448     // 14 * 32

// ---------------------------------------------------------------------------
// TDM: issue a 2-D tile load (row-major, 2-byte elements) into LDS.
// tile: tile_d1 rows x tile_d0 elems, row stride = stride_elems.
// Must be executed by exactly one wave (EXEC ignored, per-wave TENSORcnt).
// ---------------------------------------------------------------------------
__device__ __forceinline__ void tdm_load_2d(unsigned lds_addr, const void* gptr,
                                            int tensor_d0, int tensor_d1,
                                            int tile_d0, int tile_d1,
                                            int stride_elems) {
  unsigned long long ga = (unsigned long long)gptr;
  u32x4 g0;
  g0[0] = 1u;                                   // count=1, user descriptor
  g0[1] = lds_addr;                             // LDS byte address
  g0[2] = (unsigned)ga;                         // global_addr[31:0]
  g0[3] = (unsigned)(ga >> 32) | (2u << 30);    // global_addr[56:32] | type=2
  i32x8 g1;
  g1[0] = (1 << 16);                                            // data_size=2B
  g1[1] = (int)(((unsigned)tensor_d0 & 0xFFFFu) << 16);         // dim0 lo16
  g1[2] = (int)(((unsigned)tensor_d0 >> 16) |
                (((unsigned)tensor_d1 & 0xFFFFu) << 16));       // dim0 hi | dim1 lo
  g1[3] = (int)((((unsigned)tensor_d1 >> 16) & 0xFFFFu) |
                ((unsigned)tile_d0 << 16));                     // dim1 hi | tile_dim0
  g1[4] = tile_d1;                                              // tile_dim1 (dim2=0)
  g1[5] = stride_elems;                                         // dim0_stride lo32
  g1[6] = 0;
  g1[7] = 0;
  i32x4 z4 = {0, 0, 0, 0};
#if __clang_major__ >= 23
  i32x8 z8 = {0, 0, 0, 0, 0, 0, 0, 0};
  __builtin_amdgcn_tensor_load_to_lds(g0, g1, z4, z4, z8, 0);
#else
  __builtin_amdgcn_tensor_load_to_lds(g0, g1, z4, z4, 0);
#endif
}

// ---------------------------------------------------------------------------
// f32 -> bf16 conversion (plain, for nothing currently) and transposing form
// dst[n*K + k] = src[k*N + n]   (weights are small; done once per call)
// ---------------------------------------------------------------------------
__global__ __launch_bounds__(256) void cvt_bf16_t_kernel(const float* __restrict__ src,
                                                         bf16* __restrict__ dst,
                                                         int K, int N) {
  int i = blockIdx.x * 256 + threadIdx.x;
  if (i < K * N) {
    int n = i / K, k = i % K;
    dst[i] = (bf16)src[(size_t)k * N + n];
  }
}

// ---------------------------------------------------------------------------
// LayerNorm over channels, input NCHW f32 -> xn [t][C] f32 + bf16
// ---------------------------------------------------------------------------
__global__ __launch_bounds__(256) void ln1_kernel(const float* __restrict__ x,
                                                  const float* __restrict__ g,
                                                  const float* __restrict__ b,
                                                  float* __restrict__ xnf,
                                                  bf16* __restrict__ xnb) {
  int t = blockIdx.x * 256 + threadIdx.x;
  float vals[CCH];
  float mean = 0.f;
#pragma unroll
  for (int c = 0; c < CCH; ++c) { vals[c] = x[(size_t)c * NT + t]; mean += vals[c]; }
  mean *= (1.f / CCH);
  float var = 0.f;
#pragma unroll
  for (int c = 0; c < CCH; ++c) { float d = vals[c] - mean; var += d * d; }
  var *= (1.f / CCH);
  float inv = rsqrtf(var + 1e-5f);
#pragma unroll
  for (int c = 0; c < CCH; ++c) {
    float v = (vals[c] - mean) * inv * g[c] + b[c];
    xnf[(size_t)t * CCH + c] = v;
    xnb[(size_t)t * CCH + c] = (bf16)v;
  }
}

// LayerNorm over row-major [t][C] f32 -> bf16
__global__ __launch_bounds__(256) void ln2_kernel(const float* __restrict__ y,
                                                  const float* __restrict__ g,
                                                  const float* __restrict__ b,
                                                  bf16* __restrict__ h) {
  int t = blockIdx.x * 256 + threadIdx.x;
  const float* row = y + (size_t)t * CCH;
  float mean = 0.f;
#pragma unroll 8
  for (int c = 0; c < CCH; ++c) mean += row[c];
  mean *= (1.f / CCH);
  float var = 0.f;
#pragma unroll 8
  for (int c = 0; c < CCH; ++c) { float d = row[c] - mean; var += d * d; }
  var *= (1.f / CCH);
  float inv = rsqrtf(var + 1e-5f);
#pragma unroll 8
  for (int c = 0; c < CCH; ++c)
    h[(size_t)t * CCH + c] = (bf16)((row[c] - mean) * inv * g[c] + b[c]);
}

// ---------------------------------------------------------------------------
// bf16 WMMA GEMM with TDM double-buffered LDS staging.
// A[M,K] row-major, BT[N,K] row-major (pre-transposed weights).
// block = 256 threads = 8 waves; tile 128x64; BK=32.
// Wave wv owns M-tile wv (16 rows) x 4 N-tiles -> 4 wmma per K-step.
// out = act(A@B + bias) (+ res); outputs f32 and/or bf16.
// ---------------------------------------------------------------------------
__global__ __launch_bounds__(256) void gemm_bf16_wmma(const bf16* __restrict__ A,
                                                      const bf16* __restrict__ BT,
                                                      const float* __restrict__ bias,
                                                      const float* __restrict__ res,
                                                      float* __restrict__ outf,
                                                      bf16* __restrict__ outb,
                                                      int M, int N, int K, int gelu) {
  __shared__ bf16 As[2][128][32];
  __shared__ bf16 Bs[2][64][32];
  const int tid = threadIdx.x;
  const int lane = tid & 31, wv = tid >> 5;
  const int hl = lane >> 4, ll = lane & 15;
  const int bm = blockIdx.y, bn = blockIdx.x;
  const int nsteps = K >> 5;
  v8f acc[4] = {{}, {}, {}, {}};

  const bf16* Abase = A + (size_t)bm * 128 * K;
  const bf16* Bbase = BT + (size_t)bn * 64 * K;

  if (wv == 0) {
    tdm_load_2d((unsigned)(uintptr_t)&As[0][0][0], Abase, K, M, 32, 128, K);
    tdm_load_2d((unsigned)(uintptr_t)&Bs[0][0][0], Bbase, K, N, 32, 64, K);
  }
  for (int step = 0; step < nsteps; ++step) {
    const int buf = step & 1;
    if (wv == 0) __builtin_amdgcn_s_wait_tensorcnt(0);
    __syncthreads();  // current buffer ready for all waves; prior reads done
    if (wv == 0 && step + 1 < nsteps) {
      int k0 = (step + 1) * 32;
      tdm_load_2d((unsigned)(uintptr_t)&As[buf ^ 1][0][0], Abase + k0, K, M, 32, 128, K);
      tdm_load_2d((unsigned)(uintptr_t)&Bs[buf ^ 1][0][0], Bbase + k0, K, N, 32, 64, K);
    }
    v16bf a;
#pragma unroll
    for (int j = 0; j < 8; ++j) a[j] = As[buf][wv * 16 + ll][hl * 8 + j];
#pragma unroll
    for (int j = 0; j < 8; ++j) a[8 + j] = As[buf][wv * 16 + ll][16 + hl * 8 + j];
#pragma unroll
    for (int nt = 0; nt < 4; ++nt) {
      v16bf b;
#pragma unroll
      for (int j = 0; j < 16; ++j) b[j] = Bs[buf][nt * 16 + ll][(j < 8) ? (hl * 8 + j) : (16 + hl * 8 + j - 8)];
      acc[nt] = __builtin_amdgcn_wmma_f32_16x16x32_bf16(false, a, false, b, (short)0,
                                                        acc[nt], false, false);
    }
  }
#pragma unroll
  for (int nt = 0; nt < 4; ++nt) {
    int gn = bn * 64 + nt * 16 + ll;
    float bi = bias[gn];
#pragma unroll
    for (int r = 0; r < 8; ++r) {
      int gm = bm * 128 + wv * 16 + hl * 8 + r;
      float v = acc[nt][r] + bi;
      if (gelu) {
        float x3 = v * v * v;
        v = 0.5f * v * (1.0f + tanhf(0.7978845608f * (v + 0.044715f * x3)));
      }
      if (res)  v += res[(size_t)gm * N + gn];
      if (outf) outf[(size_t)gm * N + gn] = v;
      if (outb) outb[(size_t)gm * N + gn] = (bf16)v;
    }
  }
}

// ---------------------------------------------------------------------------
// Focal pooling: level1 (12x12, p=16), level2 (6x6, p=32); k/v means -> bf16
// ---------------------------------------------------------------------------
__global__ __launch_bounds__(128) void pool_kernel(const bf16* __restrict__ qkv,
                                                   bf16* __restrict__ k1, bf16* __restrict__ v1,
                                                   bf16* __restrict__ k2, bf16* __restrict__ v2) {
  int cell = blockIdx.x;   // 0..143 level1, 144..179 level2
  int c = threadIdx.x;     // channel
  int p, py, px;
  if (cell < 144) { p = 16; py = cell / 12; px = cell % 12; }
  else            { p = 32; int cc = cell - 144; py = cc / 6; px = cc % 6; }
  float sk = 0.f, sv = 0.f;
  for (int i = 0; i < p; ++i)
    for (int j = 0; j < p; ++j) {
      int token = (py * p + i) * IMG + px * p + j;
      sk += (float)qkv[(size_t)token * 384 + 128 + c];
      sv += (float)qkv[(size_t)token * 384 + 256 + c];
    }
  float inv = 1.f / (float)(p * p);
  if (cell < 144) { k1[cell * CCH + c] = (bf16)(sk * inv); v1[cell * CCH + c] = (bf16)(sv * inv); }
  else { int cc = cell - 144; k2[cc * CCH + c] = (bf16)(sk * inv); v2[cc * CCH + c] = (bf16)(sv * inv); }
}

// ---------------------------------------------------------------------------
// Focal window attention, flash-style online softmax, WMMA bf16.
// grid = (8 heads, 144 windows), block = 256 (8 waves).
// ---------------------------------------------------------------------------
__global__ __launch_bounds__(256) void attn_kernel(const bf16* __restrict__ qkv,
                                                   const bf16* __restrict__ k1,
                                                   const bf16* __restrict__ v1,
                                                   const bf16* __restrict__ k2,
                                                   const bf16* __restrict__ v2,
                                                   bf16* __restrict__ o) {
  __shared__ bf16  Ksm[LPAD][16];       // [key][hd]
  __shared__ bf16  Vsm[16][LPAD];       // [hd][key]
  __shared__ float Ssm[8][16][32];      // per-wave score staging (C->A transpose)
  __shared__ float rmax[8][16], rsum[8][16], rscale[8][16];

  const int head = blockIdx.x;
  const int wix = blockIdx.y;
  const int wy = wix / 12, wx = wix % 12;
  const int tid = threadIdx.x, lane = tid & 31, wv = tid >> 5;
  const int hl = lane >> 4, ll = lane & 15;

  for (int kk = tid; kk < LPAD; kk += 256) {
    const bf16 *kp = nullptr, *vp = nullptr;
    if (kk < 256) {
      int token = (wy * 16 + (kk >> 4)) * IMG + wx * 16 + (kk & 15);
      kp = qkv + (size_t)token * 384 + 128 + head * 16;
      vp = qkv + (size_t)token * 384 + 256 + head * 16;
    } else if (kk < 400) {
      int idx = kk - 256;
      kp = k1 + idx * CCH + head * 16;
      vp = v1 + idx * CCH + head * 16;
    } else if (kk < LTOT) {
      int idx = kk - 400;
      kp = k2 + idx * CCH + head * 16;
      vp = v2 + idx * CCH + head * 16;
    }
#pragma unroll
    for (int d = 0; d < 16; ++d) {
      Ksm[kk][d] = kp ? kp[d] : (bf16)0.f;
      Vsm[d][kk] = vp ? vp[d] : (bf16)0.f;
    }
  }
  __syncthreads();

  const float scale = 0.25f; // 1/sqrt(16)
  for (int half = 0; half < 2; ++half) {
    int mt = wv + half * 8;
    int q = mt * 16 + ll;
    int token = (wy * 16 + (q >> 4)) * IMG + wx * 16 + (q & 15);
    const bf16* qp = qkv + (size_t)token * 384 + head * 16 + hl * 8;
    v16bf aq;
#pragma unroll
    for (int j = 0; j < 8; ++j) aq[j] = qp[j];
#pragma unroll
    for (int j = 8; j < 16; ++j) aq[j] = (bf16)0.f;

    if (lane < 16) { rmax[wv][lane] = -3.0e38f; rsum[wv][lane] = 0.f; }
    asm volatile("s_wait_dscnt 0x0" ::: "memory");

    v8f o_acc = {};
    for (int c = 0; c < 14; ++c) {
#pragma unroll
      for (int t2 = 0; t2 < 2; ++t2) {
        int keyb = c * 32 + t2 * 16;
        v16bf bk;
#pragma unroll
        for (int j = 0; j < 16; ++j)
          bk[j] = (j < 8) ? Ksm[keyb + ll][hl * 8 + j] : (bf16)0.f;
        v8f zero = {};
        v8f s = __builtin_amdgcn_wmma_f32_16x16x32_bf16(false, aq, false, bk, (short)0,
                                                        zero, false, false);
        int key = keyb + ll;
#pragma unroll
        for (int r = 0; r < 8; ++r) {
          float val = s[r] * scale;
          if (key >= LTOT) val = -3.0e38f;
          Ssm[wv][r + hl * 8][t2 * 16 + ll] = val;
        }
      }
      asm volatile("s_wait_dscnt 0x0" ::: "memory");
      if (lane < 16) {
        float m = rmax[wv][lane], mold = m;
#pragma unroll
        for (int j = 0; j < 32; ++j) m = fmaxf(m, Ssm[wv][lane][j]);
        float f = __expf(mold - m);
        float sum = 0.f;
#pragma unroll
        for (int j = 0; j < 32; ++j) {
          float p = __expf(Ssm[wv][lane][j] - m);
          Ssm[wv][lane][j] = p;
          sum += p;
        }
        rmax[wv][lane] = m;
        rsum[wv][lane] = rsum[wv][lane] * f + sum;
        rscale[wv][lane] = f;
      }
      asm volatile("s_wait_dscnt 0x0" ::: "memory");
#pragma unroll
      for (int r = 0; r < 8; ++r) o_acc[r] *= rscale[wv][r + hl * 8];
      v16bf ap, bv;
#pragma unroll
      for (int j = 0; j < 8; ++j) ap[j] = (bf16)Ssm[wv][ll][hl * 8 + j];
#pragma unroll
      for (int j = 0; j < 8; ++j) ap[8 + j] = (bf16)Ssm[wv][ll][16 + hl * 8 + j];
#pragma unroll
      for (int j = 0; j < 16; ++j) bv[j] = Vsm[ll][c * 32 + ((j < 8) ? (hl * 8 + j) : (16 + hl * 8 + j - 8))];
      o_acc = __builtin_amdgcn_wmma_f32_16x16x32_bf16(false, ap, false, bv, (short)0,
                                                      o_acc, false, false);
    }
#pragma unroll
    for (int r = 0; r < 8; ++r) {
      int row = r + hl * 8;
      float val = o_acc[r] / rsum[wv][row];
      int qi = mt * 16 + row;
      int tok = (wy * 16 + (qi >> 4)) * IMG + wx * 16 + (qi & 15);
      o[(size_t)tok * CCH + head * 16 + ll] = (bf16)val;
    }
    asm volatile("s_wait_dscnt 0x0" ::: "memory");
  }
}

// out(NCHW) = x + y2^T
__global__ __launch_bounds__(256) void final_kernel(const float* __restrict__ x,
                                                    const float* __restrict__ y2,
                                                    float* __restrict__ out) {
  int idx = blockIdx.x * 256 + threadIdx.x;
  int c = idx / NT, t = idx % NT;
  out[idx] = x[idx] + y2[(size_t)t * CCH + c];
}

// ---------------------------------------------------------------------------
extern "C" void kernel_launch(void* const* d_in, const int* in_sizes, int n_in,
                              void* d_out, int out_size, void* d_ws, size_t ws_size,
                              hipStream_t stream) {
  (void)in_sizes; (void)n_in; (void)out_size; (void)ws_size;
  const float* x      = (const float*)d_in[0];
  const float* ln1_g  = (const float*)d_in[1];
  const float* ln1_b  = (const float*)d_in[2];
  const float* qkv_w  = (const float*)d_in[3];
  const float* qkv_b  = (const float*)d_in[4];
  const float* proj_w = (const float*)d_in[5];
  const float* proj_b = (const float*)d_in[6];
  const float* ln2_g  = (const float*)d_in[7];
  const float* ln2_b  = (const float*)d_in[8];
  const float* mlp_w1 = (const float*)d_in[9];
  const float* mlp_b1 = (const float*)d_in[10];
  const float* mlp_w2 = (const float*)d_in[11];
  const float* mlp_b2 = (const float*)d_in[12];

  char* ws = (char*)d_ws;
  size_t o_xnf = 0;
  size_t o_xnb = o_xnf + (size_t)NT * 128 * 4;
  size_t o_qkv = o_xnb + (size_t)NT * 128 * 2;
  size_t o_ob  = o_qkv + (size_t)NT * 384 * 2;
  size_t o_y   = o_ob  + (size_t)NT * 128 * 2;
  size_t o_hb  = o_y   + (size_t)NT * 128 * 4;
  size_t o_hid = o_hb  + (size_t)NT * 128 * 2;
  size_t o_wq  = o_hid + (size_t)NT * 512 * 2;
  size_t o_wp  = o_wq  + 128 * 384 * 2;
  size_t o_w1  = o_wp  + 128 * 128 * 2;
  size_t o_w2  = o_w1  + 128 * 512 * 2;
  size_t o_k1  = o_w2  + 512 * 128 * 2;
  size_t o_v1  = o_k1  + 144 * 128 * 2;
  size_t o_k2  = o_v1  + 144 * 128 * 2;
  size_t o_v2  = o_k2  + 36 * 128 * 2;
  size_t o_y2  = o_xnf; // xn_f32 dead after proj residual -> alias

  float* xnf  = (float*)(ws + o_xnf);
  bf16*  xnb  = (bf16*)(ws + o_xnb);
  bf16*  qkvb = (bf16*)(ws + o_qkv);
  bf16*  ob   = (bf16*)(ws + o_ob);
  float* y    = (float*)(ws + o_y);
  bf16*  hb   = (bf16*)(ws + o_hb);
  bf16*  hid  = (bf16*)(ws + o_hid);
  bf16*  wq   = (bf16*)(ws + o_wq);   // [384][128]  (N x K)
  bf16*  wp   = (bf16*)(ws + o_wp);   // [128][128]
  bf16*  w1   = (bf16*)(ws + o_w1);   // [512][128]
  bf16*  w2   = (bf16*)(ws + o_w2);   // [128][512]
  bf16*  k1   = (bf16*)(ws + o_k1);
  bf16*  v1   = (bf16*)(ws + o_v1);
  bf16*  k2   = (bf16*)(ws + o_k2);
  bf16*  v2   = (bf16*)(ws + o_v2);
  float* y2   = (float*)(ws + o_y2);

  // weights -> bf16, transposed to [N][K]
  cvt_bf16_t_kernel<<<(128 * 384 + 255) / 256, 256, 0, stream>>>(qkv_w, wq, 128, 384);
  cvt_bf16_t_kernel<<<(128 * 128 + 255) / 256, 256, 0, stream>>>(proj_w, wp, 128, 128);
  cvt_bf16_t_kernel<<<(128 * 512 + 255) / 256, 256, 0, stream>>>(mlp_w1, w1, 128, 512);
  cvt_bf16_t_kernel<<<(512 * 128 + 255) / 256, 256, 0, stream>>>(mlp_w2, w2, 512, 128);

  // LN1
  ln1_kernel<<<NT / 256, 256, 0, stream>>>(x, ln1_g, ln1_b, xnf, xnb);
  // QKV GEMM (bf16 out)
  gemm_bf16_wmma<<<dim3(384 / 64, NT / 128), 256, 0, stream>>>(
      xnb, wq, qkv_b, nullptr, nullptr, qkvb, NT, 384, 128, 0);
  // focal pooling
  pool_kernel<<<180, 128, 0, stream>>>(qkvb, k1, v1, k2, v2);
  // attention
  attn_kernel<<<dim3(8, 144), 256, 0, stream>>>(qkvb, k1, v1, k2, v2, ob);
  // proj GEMM + xn residual -> y (f32)
  gemm_bf16_wmma<<<dim3(128 / 64, NT / 128), 256, 0, stream>>>(
      ob, wp, proj_b, xnf, y, nullptr, NT, 128, 128, 0);
  // LN2
  ln2_kernel<<<NT / 256, 256, 0, stream>>>(y, ln2_g, ln2_b, hb);
  // MLP1 + GELU (bf16 out)
  gemm_bf16_wmma<<<dim3(512 / 64, NT / 128), 256, 0, stream>>>(
      hb, w1, mlp_b1, nullptr, nullptr, hid, NT, 512, 128, 1);
  // MLP2 + y residual -> y2 (f32)
  gemm_bf16_wmma<<<dim3(128 / 64, NT / 128), 256, 0, stream>>>(
      hid, w2, mlp_b2, y, y2, nullptr, NT, 128, 512, 0);
  // outer residual, back to NCHW
  final_kernel<<<(128 * NT) / 256, 256, 0, stream>>>(x, y2, (float*)d_out);
}
